// KANLayerFast_66821101191171
// MI455X (gfx1250) — matrix operand order repaired and checked by reading, compiled, not run on hardware
//
#include <hip/hip_runtime.h>
#include <hip/hip_bf16.h>

// ---------------- problem constants (match reference) ----------------
constexpr int KD_IN  = 256;
constexpr int KD_OUT = 512;
constexpr int KKNOT  = 64;      // K knots
constexpr int KCAT   = 128;     // 64 y-slots + 64 d-slots
#define EPSF 1e-12f

// ---------------- CDNA5 vector types ----------------
typedef __attribute__((ext_vector_type(16))) __bf16 bf16x16;
typedef __attribute__((ext_vector_type(8)))  float  f32x8;
typedef __attribute__((ext_vector_type(4)))  int    int4v;

typedef __attribute__((address_space(1))) int4v g_int4v;   // global int4
typedef __attribute__((address_space(3))) int4v l_int4v;   // LDS int4

union Frag16 { uint4 q[2]; bf16x16 v; };   // 32 bytes = one WMMA bf16 A/B fragment

// async global->LDS path (gfx1250); fall back to plain LDS staging if absent
#if defined(__has_builtin)
#if __has_builtin(__builtin_amdgcn_global_load_async_to_lds_b128) && \
    __has_builtin(__builtin_amdgcn_s_wait_asynccnt)
#define USE_ASYNC_LDS 1
#endif
#endif
#ifndef USE_ASYNC_LDS
#define USE_ASYNC_LDS 0
#endif

__device__ __forceinline__ unsigned short f2bf(float f) {
    unsigned u = __float_as_uint(f);
    unsigned r = u + 0x7FFFu + ((u >> 16) & 1u);   // round-to-nearest-even
    return (unsigned short)(r >> 16);
}

// =====================================================================
// Kernel 1: build Mcat[i][o][0:128] = bf16(coeffs[o,i,:]) ++ bf16(pchip_d[o,i,:])
// One wave32 per (o,i) pair; lanes span the knot dimension; slopes via __shfl.
// =====================================================================
__global__ __launch_bounds__(256)
void kan_prep(const float* __restrict__ C, const float* __restrict__ knots,
              unsigned short* __restrict__ MC)
{
    const int wave = threadIdx.x >> 5;
    const int lane = threadIdx.x & 31;
    const int pair = blockIdx.x * 8 + wave;       // 0 .. 512*256-1
    const int i = pair >> 9;                      // /512
    const int o = pair & 511;

    const float* y = C + ((size_t)o * KD_IN + i) * KKNOT;
    unsigned short* mrow = MC + ((size_t)i * KD_OUT + o) * KCAT;

    const float hh   = knots[1] - knots[0];       // uniform spacing
    const float invH = 1.0f / (hh + EPSF);

    // lane handles knot j=lane (half A) and j=lane+32 (half B)
    float ya = y[lane];
    float yb = y[lane + 32];

    float ya_n = __shfl_down(ya, 1);              // y[lane+1]   (lanes 0..30)
    float yb_n = __shfl_down(yb, 1);              // y[lane+33]  (lanes 0..30)
    float y32  = __shfl(yb, 0);                   // y[32]

    float da = ((lane == 31) ? (y32 - ya) : (ya_n - ya)) * invH; // delta_j, j=0..31
    float db = (yb_n - yb) * invH;                               // delta_{j+32}, valid 32..62

    float da_p = __shfl_up(da, 1);                // delta_{j-1}
    float db_p = __shfl_up(db, 1);                // delta_{j+31}
    float d31  = __shfl(da, 31);
    if (lane == 0) db_p = d31;                    // delta_31 feeds j=32

    // broadcast values needed by endpoint formulas (computed convergently)
    float d1s = __shfl(da, 1);                    // delta_1
    float d62 = __shfl(db, 30);                   // delta_62
    float d61 = __shfl(db, 29);                   // delta_61

    auto interior = [&](float dp, float dn) {
        float w = 6.0f * hh / (3.0f * hh / (dp + EPSF) + 3.0f * hh / (dn + EPSF) + EPSF);
        return (dp * dn > 0.0f) ? w : 0.0f;
    };
    auto limit = [&](float di, float de) {
        di = (di * de <= 0.0f) ? 0.0f : di;
        return (fabsf(di) > 3.0f * fabsf(de)) ? 3.0f * de : di;
    };

    float intA = interior(da_p, da);
    float e0   = limit(0.5f * (3.0f * da - d1s), da);      // left endpoint
    float slope_a = (lane == 0) ? e0 : intA;

    float intB = interior(db_p, db);
    float eN   = limit(0.5f * (3.0f * d62 - d61), d62);    // right endpoint
    float slope_b = (lane == 31) ? eN : intB;

    mrow[lane]       = f2bf(ya);
    mrow[lane + 32]  = f2bf(yb);
    mrow[64 + lane]  = f2bf(slope_a);
    mrow[96 + lane]  = f2bf(slope_b);
}

// =====================================================================
// Kernel 2: fused spline-activation x Mcat GEMM via bf16 WMMA.
//   out[b,o] = sum_{i,kk} W[b,i,kk] * Mcat[o,i,kk] + bias[o]
// BM=128 x BN=128 tile per workgroup; wave tile 64x32 (8 accumulators);
// double-buffered LDS so async staging + spline W-build overlap the WMMAs.
// =====================================================================
constexpr int BM  = 128;
constexpr int BN  = 128;
constexpr int LDW = 136;   // padded LDS row stride (elements) -> 272B, staggers banks

__global__ __launch_bounds__(256)
void kan_gemm(const float* __restrict__ X, const unsigned short* __restrict__ MC,
              const float* __restrict__ bias, const float* __restrict__ knots,
              float* __restrict__ out)
{
    __shared__ __align__(16) unsigned short sW[2][BM * LDW];  // activation tiles
    __shared__ __align__(16) unsigned short sM[2][BN * LDW];  // Mcat tiles

    const int tid  = threadIdx.x;
    const int lane = tid & 31;
    const int wave = tid >> 5;
    const int wr   = wave >> 2;        // 0..1  -> 64-row band
    const int wc   = wave & 3;         // 0..3  -> 32-col band (2 x 16 subtiles)
    const int bm0  = blockIdx.x * BM;
    const int on0  = blockIdx.y * BN;

    // spline-build mapping: 2 threads per batch row (one per 64-slot half)
    const int srow  = tid >> 1;
    const int shalf = tid & 1;

    const float kmin  = knots[0];
    const float kmax  = knots[KKNOT - 1];
    const float hh    = (kmax - kmin) / (float)(KKNOT - 1);
    const float invh  = (float)(KKNOT - 1) / (kmax - kmin);
    const float hcell = hh + EPSF;

    // ---- stage Mcat tile: 128 o-rows x 128 kk (16 x 16B chunks per row) ----
    auto stage_sM = [&](int i, int buf) {
        const unsigned short* gbase = MC + (size_t)(i * KD_OUT + on0) * KCAT;
#pragma unroll
        for (int j = 0; j < 8; ++j) {
            int c    = tid + 256 * j;              // 0..2047 chunks of 16B
            int orow = c >> 4;                     // 128 elements = 16 chunks/row
            int sub  = c & 15;
            int loff = orow * LDW + sub * 8;
            const unsigned short* src = gbase + orow * KCAT + sub * 8;
#if USE_ASYNC_LDS
            __builtin_amdgcn_global_load_async_to_lds_b128(
                (g_int4v*)src, (l_int4v*)&sM[buf][loff], 0, 0);
#else
            *(uint4*)&sM[buf][loff] = *(const uint4*)src;
#endif
        }
    };

    // ---- build sparse->dense activation half-row in LDS ----
    auto build_sW = [&](int i, int buf) {
        float xv = X[(size_t)(bm0 + srow) * KD_IN + i];
        float xc = fminf(fmaxf(xv, kmin), kmax);
        float fi = (xc - kmin) * invh;
        int  idx = (int)fi; if (idx > KKNOT - 2) idx = KKNOT - 2;
        float x0 = kmin + (float)idx * hh;
        float t  = (xc - x0) / hcell;
        float t2 = t * t, t3 = t2 * t;
        bool below = xv < kmin, above = xv > kmax;
        bool ext   = below || above;
        int kk0 = ext ? (below ? 0 : KKNOT - 1) : idx;
        int kk1 = ext ? kk0 : idx + 1;
        float w0, w1;
        if (shalf == 0) {          // y-weights
            w0 = ext ? 1.0f : (2.0f * t3 - 3.0f * t2 + 1.0f);
            w1 = ext ? 0.0f : (-2.0f * t3 + 3.0f * t2);
        } else {                   // derivative weights (include h factor)
            w0 = ext ? (below ? xv - kmin : xv - kmax) : (t3 - 2.0f * t2 + t) * hcell;
            w1 = ext ? 0.0f : (t3 - t2) * hcell;
        }
        unsigned base = (unsigned)(srow * LDW + shalf * KKNOT);
        uint4 z; z.x = z.y = z.z = z.w = 0u;
#pragma unroll
        for (int j = 0; j < 8; ++j) *(uint4*)&sW[buf][base + j * 8] = z;
        sW[buf][base + kk0] = f2bf(w0);
        if (kk1 != kk0) sW[buf][base + kk1] = f2bf(w1);
    };

    f32x8 acc[4][2] = {};              // 4 row-subtiles x 2 col-subtiles per wave

    // prologue: fill buffer 0
    stage_sM(0, 0);
    build_sW(0, 0);
#if USE_ASYNC_LDS
    __builtin_amdgcn_s_wait_asynccnt(0);
#endif
    __syncthreads();

    for (int i = 0; i < KD_IN; ++i) {
        const int buf = i & 1;

        // kick off next iteration's staging into the other buffer; the async
        // loads and the VALU spline build overlap this iteration's WMMAs.
        if (i + 1 < KD_IN) {
            stage_sM(i + 1, buf ^ 1);
            build_sW(i + 1, buf ^ 1);
        }

        // ---- 4 K-chunks of 32: 2 B frags + 4 A frags -> 8 WMMAs ----
#pragma unroll
        for (int kc = 0; kc < 4; ++kc) {
            Frag16 bfm[2], af[4];
            int kb = kc * 32 + ((lane >> 4) << 4);         // 16 contiguous kk
#pragma unroll
            for (int n = 0; n < 2; ++n) {
                int colL = wc * 32 + n * 16 + (lane & 15);
                const unsigned short* mb = &sM[buf][colL * LDW + kb];
                bfm[n].q[0] = *(const uint4*)mb;
                bfm[n].q[1] = *(const uint4*)(mb + 8);
            }
            int ka = kc * 32 + ((lane >> 4) << 4);         // 16 contiguous kk
#pragma unroll
            for (int s = 0; s < 4; ++s) {
                int rowL = wr * 64 + s * 16 + (lane & 15);
                const unsigned short* wb = &sW[buf][rowL * LDW + ka];
                af[s].q[0] = *(const uint4*)wb;
                af[s].q[1] = *(const uint4*)(wb + 8);
            }
#pragma unroll
            for (int s = 0; s < 4; ++s) {
#pragma unroll
                for (int n = 0; n < 2; ++n) {
                    acc[s][n] = __builtin_amdgcn_wmma_f32_16x16x32_bf16(
                        false, af[s].v, false, bfm[n].v, (short)0, acc[s][n], false, false);
                }
            }
        }

#if USE_ASYNC_LDS
        if (i + 1 < KD_IN) __builtin_amdgcn_s_wait_asynccnt(0);
#endif
        __syncthreads();   // next buffer fully staged; this buffer's reads done
    }

    // ---- epilogue: C/D layout -> rows r + 8*(lane>>4), col = lane&15 -------
#pragma unroll
    for (int n = 0; n < 2; ++n) {
        int gcol = on0 + wc * 32 + n * 16 + (lane & 15);
        float bv = bias[gcol];
#pragma unroll
        for (int s = 0; s < 4; ++s) {
            int rbase = bm0 + wr * 64 + s * 16 + ((lane >> 4) << 3);
#pragma unroll
            for (int r = 0; r < 8; ++r) {
                out[(size_t)(rbase + r) * KD_OUT + gcol] = acc[s][n][r] + bv;
            }
        }
    }
}

// =====================================================================
extern "C" void kernel_launch(void* const* d_in, const int* in_sizes, int n_in,
                              void* d_out, int out_size, void* d_ws, size_t ws_size,
                              hipStream_t stream)
{
    const float* x      = (const float*)d_in[0];   // (4096, 256)
    const float* coeffs = (const float*)d_in[1];   // (512, 256, 64)
    const float* bias   = (const float*)d_in[2];   // (512,)
    const float* knots  = (const float*)d_in[3];   // (64,)
    float* out = (float*)d_out;                    // (4096, 512)

    unsigned short* MC = (unsigned short*)d_ws;    // Mcat bf16: 256*512*128*2 = 32 MB

    // Kernel 1: PCHIP slopes + bf16 pack. 512*256 pairs, 8 waves/block.
    kan_prep<<<dim3((KD_OUT * KD_IN) / 8), 256, 0, stream>>>(coeffs, knots, MC);

    // Kernel 2: fused spline-activation GEMM. 4096/128 x 512/128 tiles.
    kan_gemm<<<dim3(4096 / BM, KD_OUT / BN), 256, 0, stream>>>(x, MC, bias, knots, out);
}
